// SwinTransformerBlock_13812614824530
// MI455X (gfx1250) — matrix operand order repaired and checked by reading, not compiled, not run on hardware
//
#include <hip/hip_runtime.h>
#include <hip/hip_bf16.h>
#include <math.h>

// ---------------------------------------------------------------------------
// Swin Transformer block (shifted-window MSA + MLP) for MI455X / gfx1250.
// All GEMMs use v_wmma_f32_16x16x32_bf16. All fragment loads are 16-byte
// vectorized (global_load_b128 / ds_load_b128); weights are pre-transposed so
// B fragments are contiguous per lane. K-loops are kept rolled (#pragma
// unroll 1) so accumulators + one A/B fragment fit in VGPRs without spills.
// ---------------------------------------------------------------------------

typedef __bf16 bf16;
typedef __attribute__((ext_vector_type(16))) __bf16 v16bf;
typedef __attribute__((ext_vector_type(8)))  float  v8f;
typedef unsigned u4 __attribute__((ext_vector_type(4)));

#define WMMA_BF16(a, b, c) \
  __builtin_amdgcn_wmma_f32_16x16x32_bf16(false, (a), false, (b), (short)0, (c), false, false)

static __device__ __forceinline__ bf16 f2bf(float f) {
  unsigned u = __builtin_bit_cast(unsigned, f);
  unsigned r = u + 0x7FFFu + ((u >> 16) & 1u);   // round-to-nearest-even
  unsigned short h = (unsigned short)(r >> 16);
  return __builtin_bit_cast(bf16, h);
}

static __device__ __forceinline__ float wredsum(float v) {
#pragma unroll
  for (int off = 16; off > 0; off >>= 1) v += __shfl_xor(v, off, 32);
  return v;
}

// Two 16-byte chunks -> one 16-element bf16 fragment (forces b128 loads).
static __device__ __forceinline__ v16bf load2x8(const bf16* p0, const bf16* p1) {
  struct Pair { u4 a, b; } t;
  t.a = *(const u4*)p0;
  t.b = *(const u4*)p1;
  return __builtin_bit_cast(v16bf, t);
}

// A fragment: 16x32 bf16 tile from row-major [M][K] source (ISA 7.12.2):
// lanes 0-15: row M=lane, a[0..7]=K0..7, a[8..15]=K16..23
// lanes 16-31: row M=lane-16, a[0..7]=K8..15, a[8..15]=K24..31
static __device__ __forceinline__ v16bf load_a(const bf16* A, int lda,
                                               int mBase, int kBase, int lane) {
  int row = mBase + (lane & 15);
  int kh  = (lane >> 4) * 8;
  const bf16* p = A + (size_t)row * lda + kBase;
  return load2x8(p + kh, p + 16 + kh);
}

// B fragment (32x16) where source is row-major [N][K] (i.e. already B^T):
// lanes 0-15: col N=lane, b[0..7]=K0..7, b[8..15]=K16..23; lanes 16-31: col N=lane-16, K8..15, K24..31
static __device__ __forceinline__ v16bf load_bt(const bf16* Bm, int ldb,
                                                int kBase, int nBase, int lane) {
  int col = nBase + (lane & 15);
  int k0  = kBase + (lane >> 4) * 8;
  const bf16* p = Bm + (size_t)col * ldb + k0;
  return load2x8(p, p + 16);
}

// ---------------------------------------------------------------------------
// Kernel 0: f32 [K][N] -> bf16 [N][K] transpose precast
// ---------------------------------------------------------------------------
__global__ void k_cvt_t(const float* __restrict__ s, bf16* __restrict__ d,
                        int K, int N) {
  int i = blockIdx.x * 256 + threadIdx.x;
  if (i < K * N) {
    int k = i / N, n = i % N;
    d[(size_t)n * K + k] = f2bf(s[i]);
  }
}

// ---------------------------------------------------------------------------
// Kernel 1: LN1 + cyclic shift (-3,-3) + window partition -> xw[2048][64][192]
// ---------------------------------------------------------------------------
__global__ void k_ln_window(const float* __restrict__ q, const float* __restrict__ g,
                            const float* __restrict__ bb, bf16* __restrict__ xw) {
  int wid = threadIdx.x >> 5, lane = threadIdx.x & 31;
  int win = blockIdx.x >> 4;
  int n   = ((blockIdx.x & 15) << 2) + wid;
  bf16* dst = xw + ((size_t)win * 64 + n) * 192;
  if (n >= 49) {
#pragma unroll
    for (int i = 0; i < 6; ++i) dst[i * 32 + lane] = f2bf(0.f);
    return;
  }
  int b = win >> 8, wi = win & 255, wh = wi >> 4, ww = wi & 15;
  int r = n / 7, c = n % 7;
  int sh = (wh * 7 + r + 3) % 112;   // roll(-3): xs[h] = x[(h+3)%H]
  int sw = (ww * 7 + c + 3) % 112;
  const float* src = q + ((size_t)b * 12544 + sh * 112 + sw) * 192;
  float v[6], s = 0.f;
#pragma unroll
  for (int i = 0; i < 6; ++i) { v[i] = src[i * 32 + lane]; s += v[i]; }
  float mu = wredsum(s) * (1.f / 192.f);
  float s2 = 0.f;
#pragma unroll
  for (int i = 0; i < 6; ++i) { float d = v[i] - mu; s2 += d * d; }
  float rs = rsqrtf(wredsum(s2) * (1.f / 192.f) + 1e-5f);
#pragma unroll
  for (int i = 0; i < 6; ++i) {
    int ch = i * 32 + lane;
    dst[ch] = f2bf((v[i] - mu) * rs * g[ch] + bb[ch]);
  }
}

// ---------------------------------------------------------------------------
// Kernel 2: per-window qkv GEMM [64x192]@[192x576] + bias, q scaled.
// WqT is transposed [576][192] so B frags are 2x b128. Each 6-tile chunk is
// entirely inside one of q/k/v, so the destination select is wave-uniform.
// ---------------------------------------------------------------------------
__global__ void k_qkv(const bf16* __restrict__ xw, const bf16* __restrict__ WqT,
                      const float* __restrict__ qkvb,
                      bf16* __restrict__ qb, bf16* __restrict__ kb, bf16* __restrict__ vb) {
  __shared__ bf16 As[64 * 192];
  int win = blockIdx.x;
  int tid = threadIdx.x, wid = tid >> 5, lane = tid & 31;
  {
    const u4* src = (const u4*)(xw + (size_t)win * 64 * 192);
    u4* dl = (u4*)As;
#pragma unroll
    for (int i = 0; i < 6; ++i) dl[tid + i * 256] = src[tid + i * 256];
  }
  __syncthreads();
  int mt = wid >> 1;
#pragma unroll 1
  for (int chunk = 0; chunk < 3; ++chunk) {
    int ntc = (wid & 1) * 18 + chunk * 6;
    v8f acc[6] = {};
#pragma unroll 1
    for (int kk = 0; kk < 192; kk += 32) {
      v16bf a = load_a(As, 192, mt * 16, kk, lane);
#pragma unroll
      for (int j = 0; j < 6; ++j) {
        v16bf bfr = load_bt(WqT, 192, kk, (ntc + j) * 16, lane);
        acc[j] = WMMA_BF16(a, bfr, acc[j]);
      }
    }
    int which = ntc / 12;                    // uniform: tiles 0-11 q, 12-23 k, 24-35 v
    bf16* dst0 = which == 0 ? qb : (which == 1 ? kb : vb);
    float scale = which == 0 ? 0.17677669529663687f : 1.f;   // hd^-0.5 after bias
    int half = lane >> 4, cl = lane & 15;
#pragma unroll
    for (int j = 0; j < 6; ++j) {
      int colBase = (ntc + j) * 16;
      int cc = colBase - which * 192 + cl;   // channel within q/k/v
      int head = cc >> 5, d = cc & 31;
#pragma unroll
      for (int r = 0; r < 8; ++r) {
        int row = mt * 16 + r + half * 8;    // D layout: VGPR r -> M=r / r+8
        float val = (acc[j][r] + qkvb[colBase + cl]) * scale;
        dst0[(((size_t)win * 6 + head) * 64 + row) * 32 + d] = f2bf(val);
      }
    }
  }
}

// ---------------------------------------------------------------------------
// Kernel 3: attention per (window, head). 4 waves.
// S = q k^T (WMMA, K=32) + rel-pos bias + shift mask; softmax; O = P V (K=64).
// V is staged transposed so PV B-frags are contiguous ds_load_b128.
// ---------------------------------------------------------------------------
__global__ void k_attn(const bf16* __restrict__ qb, const bf16* __restrict__ kb,
                       const bf16* __restrict__ vb, const float* __restrict__ rpb,
                       bf16* __restrict__ ao) {
  __shared__ bf16 Qs[64 * 32], Ks[64 * 32];
  __shared__ bf16 Vt[32 * 64];               // Vt[d][k] = v[k][d]
  __shared__ float Ss[64 * 64];
  __shared__ bf16 Ps[64 * 64];
  __shared__ signed char Rr[49], Ccl[49], Gid[49];
  int blk = blockIdx.x;
  int win = blk / 6, head = blk % 6;
  int tid = threadIdx.x, wid = tid >> 5, lane = tid & 31;
  int wi = win & 255, wh = wi >> 4, ww = wi & 15;
  size_t base = ((size_t)win * 6 + head) * 64 * 32;
  if (tid < 49) {                            // per-window geometry tables
    int r = tid / 7, c = tid % 7;
    Rr[tid] = (signed char)r; Ccl[tid] = (signed char)c;
    int gh = wh * 7 + r, gw = ww * 7 + c;
    Gid[tid] = (signed char)((gh < 105 ? 0 : (gh < 109 ? 1 : 2)) * 3 +
                             (gw < 105 ? 0 : (gw < 109 ? 1 : 2)));
  }
  {
    const u4* sq = (const u4*)(qb + base);
    const u4* sk = (const u4*)(kb + base);
    u4 *dq = (u4*)Qs, *dk = (u4*)Ks;
#pragma unroll
    for (int i = 0; i < 2; ++i) {
      dq[tid + i * 128] = sq[tid + i * 128];
      dk[tid + i * 128] = sk[tid + i * 128];
    }
    const unsigned* sv = (const unsigned*)(vb + base);   // transpose V
#pragma unroll
    for (int i = 0; i < 8; ++i) {
      int idx = tid + i * 128;                           // uint index, 2 d's
      int kR = idx >> 4, d2 = (idx & 15) * 2;
      unsigned w = sv[idx];
      Vt[(d2    ) * 64 + kR] = __builtin_bit_cast(bf16, (unsigned short)(w & 0xffffu));
      Vt[(d2 + 1) * 64 + kR] = __builtin_bit_cast(bf16, (unsigned short)(w >> 16));
    }
  }
  __syncthreads();
  {
    int mt = wid;
    v16bf a = load_a(Qs, 32, mt * 16, 0, lane);
    int half = lane >> 4, cl = lane & 15;
#pragma unroll 1
    for (int nt = 0; nt < 4; ++nt) {
      v16bf bfr = load_bt(Ks, 32, 0, nt * 16, lane);     // B[d][m] = k[m][d]
      v8f acc = {};
      acc = WMMA_BF16(a, bfr, acc);
#pragma unroll
      for (int r = 0; r < 8; ++r) {
        int n = mt * 16 + r + half * 8;
        int m = nt * 16 + cl;
        float s = acc[r];
        if (n < 49 && m < 49) {
          int idx = ((int)Rr[n] - (int)Rr[m] + 6) * 13 +
                    ((int)Ccl[n] - (int)Ccl[m] + 6);
          s += rpb[idx * 6 + head];
          if (Gid[n] != Gid[m]) s -= 100.f;              // shifted-window mask
        }
        Ss[n * 64 + m] = s;
      }
    }
  }
  __syncthreads();
  if (tid < 64) {                           // softmax over valid keys only
    float* row = Ss + tid * 64;
    float mx = -1e30f;
    for (int m = 0; m < 49; ++m) mx = fmaxf(mx, row[m]);
    float sum = 0.f;
    for (int m = 0; m < 49; ++m) { float e = __expf(row[m] - mx); row[m] = e; sum += e; }
    float inv = 1.f / sum;
    bf16* pr = Ps + tid * 64;
    for (int m = 0; m < 49; ++m) pr[m] = f2bf(row[m] * inv);
    for (int m = 49; m < 64; ++m) pr[m] = f2bf(0.f);     // zero pad columns
  }
  __syncthreads();
  {
    int mt = wid;
    int half = lane >> 4, cl = lane & 15;
#pragma unroll 1
    for (int nt = 0; nt < 2; ++nt) {
      v8f acc = {};
#pragma unroll
      for (int kk = 0; kk < 64; kk += 32) {
        v16bf a   = load_a(Ps, 64, mt * 16, kk, lane);
        v16bf bfr = load_bt(Vt, 64, kk, nt * 16, lane);  // B[k][d] = Vt[d][k]
        acc = WMMA_BF16(a, bfr, acc);
      }
#pragma unroll
      for (int r = 0; r < 8; ++r) {
        int n = mt * 16 + r + half * 8;
        int d = nt * 16 + cl;
        ao[((size_t)win * 64 + n) * 192 + head * 32 + d] = f2bf(acc[r]);
      }
    }
  }
}

// ---------------------------------------------------------------------------
// Kernel 4: proj GEMM [64x192]@[192x192] + bias, window-reverse + un-shift,
// residual with query -> x1 (f32). WpT transposed [192][192].
// ---------------------------------------------------------------------------
__global__ void k_proj(const bf16* __restrict__ ao, const bf16* __restrict__ WpT,
                       const float* __restrict__ pb, const float* __restrict__ query,
                       float* __restrict__ x1) {
  __shared__ bf16 As[64 * 192];
  int win = blockIdx.x;
  int tid = threadIdx.x, wid = tid >> 5, lane = tid & 31;
  {
    const u4* src = (const u4*)(ao + (size_t)win * 64 * 192);
    u4* dl = (u4*)As;
#pragma unroll
    for (int i = 0; i < 6; ++i) dl[tid + i * 256] = src[tid + i * 256];
  }
  __syncthreads();
  int b = win >> 8, wi = win & 255, wh = wi >> 4, ww = wi & 15;
  int mt = wid >> 1, ntc = (wid & 1) * 6;
  v8f acc[6] = {};
#pragma unroll 1
  for (int kk = 0; kk < 192; kk += 32) {
    v16bf a = load_a(As, 192, mt * 16, kk, lane);
#pragma unroll
    for (int j = 0; j < 6; ++j) {
      v16bf bfr = load_bt(WpT, 192, kk, (ntc + j) * 16, lane);
      acc[j] = WMMA_BF16(a, bfr, acc[j]);
    }
  }
  int half = lane >> 4, cl = lane & 15;
#pragma unroll
  for (int j = 0; j < 6; ++j) {
#pragma unroll
    for (int r = 0; r < 8; ++r) {
      int n = mt * 16 + r + half * 8;
      if (n >= 49) continue;               // drop pad rows
      int col = (ntc + j) * 16 + cl;
      int rr = n / 7, cc2 = n % 7;
      int oh = (wh * 7 + rr + 3) % 112;    // roll(+3): xs[g] -> x[(g+3)%H]
      int ow = (ww * 7 + cc2 + 3) % 112;
      size_t off = ((size_t)b * 12544 + (size_t)oh * 112 + ow) * 192 + col;
      x1[off] = acc[j][r] + pb[col] + query[off];
    }
  }
}

// ---------------------------------------------------------------------------
// Kernel 5: LN2 -> bf16
// ---------------------------------------------------------------------------
__global__ void k_ln2(const float* __restrict__ x1, const float* __restrict__ g,
                      const float* __restrict__ bb, bf16* __restrict__ hl) {
  int wid = threadIdx.x >> 5, lane = threadIdx.x & 31;
  size_t row = (size_t)blockIdx.x * 4 + wid;
  const float* src = x1 + row * 192;
  bf16* dst = hl + row * 192;
  float v[6], s = 0.f;
#pragma unroll
  for (int i = 0; i < 6; ++i) { v[i] = src[i * 32 + lane]; s += v[i]; }
  float mu = wredsum(s) * (1.f / 192.f);
  float s2 = 0.f;
#pragma unroll
  for (int i = 0; i < 6; ++i) { float d = v[i] - mu; s2 += d * d; }
  float rs = rsqrtf(wredsum(s2) * (1.f / 192.f) + 1e-5f);
#pragma unroll
  for (int i = 0; i < 6; ++i) {
    int ch = i * 32 + lane;
    dst[ch] = f2bf((v[i] - mu) * rs * g[ch] + bb[ch]);
  }
}

// ---------------------------------------------------------------------------
// Kernel 6: fc1 GEMM [100352x192]@[192x768] + bias + exact GELU -> bf16
// W1T transposed [768][192]. Grid (1568, 3).
// ---------------------------------------------------------------------------
__global__ void k_fc1(const bf16* __restrict__ hl, const bf16* __restrict__ W1T,
                      const float* __restrict__ b1, bf16* __restrict__ h1) {
  __shared__ bf16 As[64 * 192];
  int tid = threadIdx.x, wid = tid >> 5, lane = tid & 31;
  size_t rowBase = (size_t)blockIdx.x * 64;
  int colBase = blockIdx.y * 256;
  {
    const u4* src = (const u4*)(hl + rowBase * 192);
    u4* dl = (u4*)As;
#pragma unroll
    for (int i = 0; i < 6; ++i) dl[tid + i * 256] = src[tid + i * 256];
  }
  __builtin_prefetch(W1T + (size_t)(colBase + (tid & 63)) * 192, 0, 1);
  __syncthreads();
  int mt = wid >> 1, ntb = (wid & 1) * 8;
  v8f acc[8] = {};
#pragma unroll 1
  for (int kk = 0; kk < 192; kk += 32) {
    v16bf a = load_a(As, 192, mt * 16, kk, lane);
#pragma unroll
    for (int j = 0; j < 8; ++j) {
      v16bf bfr = load_bt(W1T, 192, kk, colBase + (ntb + j) * 16, lane);
      acc[j] = WMMA_BF16(a, bfr, acc[j]);
    }
  }
  int half = lane >> 4, cl = lane & 15;
#pragma unroll
  for (int j = 0; j < 8; ++j) {
#pragma unroll
    for (int r = 0; r < 8; ++r) {
      size_t row = rowBase + mt * 16 + r + half * 8;
      int col = colBase + (ntb + j) * 16 + cl;
      float y = acc[j][r] + b1[col];
      float ge = 0.5f * y * (1.f + erff(y * 0.70710678118654752f));
      h1[row * 768 + col] = f2bf(ge);
    }
  }
}

// ---------------------------------------------------------------------------
// Kernel 7: fc2 GEMM [100352x768]@[768x192] + bias + residual -> f32 out.
// W2T transposed [192][768]. K staged through LDS in 192-wide chunks.
// ---------------------------------------------------------------------------
__global__ void k_fc2(const bf16* __restrict__ h1, const bf16* __restrict__ W2T,
                      const float* __restrict__ b2, const float* __restrict__ x1,
                      float* __restrict__ out) {
  __shared__ bf16 As[64 * 192];
  int tid = threadIdx.x, wid = tid >> 5, lane = tid & 31;
  size_t rowBase = (size_t)blockIdx.x * 64;
  int mt = wid >> 1, ntb = (wid & 1) * 6;
  v8f acc[6] = {};
#pragma unroll 1
  for (int kc = 0; kc < 4; ++kc) {
    u4* dl = (u4*)As;
    const u4* sl = (const u4*)h1;                  // h1 row = 96 u4
#pragma unroll
    for (int i = 0; i < 6; ++i) {
      int idx = tid + i * 256;                     // 0..1535
      int r = idx / 24, c4 = idx % 24;
      dl[idx] = sl[(rowBase + r) * 96 + kc * 24 + c4];
    }
    if (kc < 3)                                    // prefetch next K chunk
      __builtin_prefetch(h1 + (rowBase + (tid >> 2)) * 768 + (kc + 1) * 192, 0, 1);
    __syncthreads();
#pragma unroll 1
    for (int kk = 0; kk < 192; kk += 32) {
      v16bf a = load_a(As, 192, mt * 16, kk, lane);
#pragma unroll
      for (int j = 0; j < 6; ++j) {
        v16bf bfr = load_bt(W2T, 768, kc * 192 + kk, (ntb + j) * 16, lane);
        acc[j] = WMMA_BF16(a, bfr, acc[j]);
      }
    }
    __syncthreads();
  }
  int half = lane >> 4, cl = lane & 15;
#pragma unroll
  for (int j = 0; j < 6; ++j) {
#pragma unroll
    for (int r = 0; r < 8; ++r) {
      size_t row = rowBase + mt * 16 + r + half * 8;
      int col = (ntb + j) * 16 + cl;
      size_t off = row * 192 + col;
      out[off] = acc[j][r] + b2[col] + x1[off];
    }
  }
}

// ---------------------------------------------------------------------------
extern "C" void kernel_launch(void* const* d_in, const int* in_sizes, int n_in,
                              void* d_out, int out_size, void* d_ws, size_t ws_size,
                              hipStream_t stream) {
  (void)in_sizes; (void)n_in; (void)out_size; (void)ws_size;
  const float* query = (const float*)d_in[0];
  const float* n1g   = (const float*)d_in[1];
  const float* n1b   = (const float*)d_in[2];
  const float* qkvw  = (const float*)d_in[3];
  const float* qkvb  = (const float*)d_in[4];
  const float* rpb   = (const float*)d_in[5];
  const float* projw = (const float*)d_in[6];
  const float* projb = (const float*)d_in[7];
  const float* n2g   = (const float*)d_in[8];
  const float* n2b   = (const float*)d_in[9];
  const float* fc1w  = (const float*)d_in[10];
  const float* fc1b  = (const float*)d_in[11];
  const float* fc2w  = (const float*)d_in[12];
  const float* fc2b  = (const float*)d_in[13];

  char* ws = (char*)d_ws;
  size_t off = 0;
  auto take = [&](size_t bytes) -> char* {
    char* p = ws + off;
    off += (bytes + 255) & ~(size_t)255;
    return p;
  };
  bf16* WqkvT = (bf16*)take((size_t)576 * 192 * 2);   // [N][K]
  bf16* WprojT= (bf16*)take((size_t)192 * 192 * 2);
  bf16* W1T   = (bf16*)take((size_t)768 * 192 * 2);
  bf16* W2T   = (bf16*)take((size_t)192 * 768 * 2);
  bf16* xw    = (bf16*)take((size_t)2048 * 64 * 192 * 2);
  bf16* qb    = (bf16*)take((size_t)2048 * 6 * 64 * 32 * 2);
  bf16* kbf   = (bf16*)take((size_t)2048 * 6 * 64 * 32 * 2);
  bf16* vbf   = (bf16*)take((size_t)2048 * 6 * 64 * 32 * 2);
  bf16* ao    = (bf16*)take((size_t)2048 * 64 * 192 * 2);
  float* x1   = (float*)take((size_t)100352 * 192 * 4);
  bf16* hl    = (bf16*)take((size_t)100352 * 192 * 2);
  bf16* h1    = (bf16*)take((size_t)100352 * 768 * 2);

  k_cvt_t<<<(110592 + 255) / 256, 256, 0, stream>>>(qkvw, WqkvT, 192, 576);
  k_cvt_t<<<(36864  + 255) / 256, 256, 0, stream>>>(projw, WprojT, 192, 192);
  k_cvt_t<<<(147456 + 255) / 256, 256, 0, stream>>>(fc1w, W1T, 192, 768);
  k_cvt_t<<<(147456 + 255) / 256, 256, 0, stream>>>(fc2w, W2T, 768, 192);

  k_ln_window<<<2048 * 16, 128, 0, stream>>>(query, n1g, n1b, xw);
  k_qkv<<<2048, 256, 0, stream>>>(xw, WqkvT, qkvb, qb, kbf, vbf);
  k_attn<<<2048 * 6, 128, 0, stream>>>(qb, kbf, vbf, rpb, ao);
  k_proj<<<2048, 256, 0, stream>>>(ao, WprojT, projb, query, x1);
  k_ln2<<<100352 / 4, 128, 0, stream>>>(x1, n2g, n2b, hl);
  dim3 g6(1568, 3);
  k_fc1<<<g6, 256, 0, stream>>>(hl, W1T, fc1b, h1);
  k_fc2<<<1568, 256, 0, stream>>>(h1, W2T, fc2b, x1, (float*)d_out);
}
